// NeuralMemoryBlock_53042846106013
// MI455X (gfx1250) — matrix operand order repaired and checked by reading, not compiled, not run on hardware
//
#include <hip/hip_runtime.h>
#include <hip/hip_bf16.h>

#define DIMX   512
#define HEADSX 4
#define DHEADX 128
#define CHUNKX 64
#define NKVX   2
#define BBX    4
#define SSX    8192
#define NTOK   (BBX*SSX)      /* 32768 tokens */
#define NCHUNK (SSX/CHUNKX)   /* 128 chunks */
#define KQKV   512
#define NQKV   2560

typedef __bf16 bf16_t;
typedef __bf16 v16bf __attribute__((ext_vector_type(16)));
typedef __bf16 v8bf  __attribute__((ext_vector_type(8)));
typedef float  v8f   __attribute__((ext_vector_type(8)));

__device__ __forceinline__ float bf2f(bf16_t x) {
  unsigned short u = __builtin_bit_cast(unsigned short, x);
  unsigned int v = ((unsigned int)u) << 16;
  return __builtin_bit_cast(float, v);
}
__device__ __forceinline__ bf16_t f2bf(float f) {
  unsigned int u = __builtin_bit_cast(unsigned int, f);
  u += 0x7fffu + ((u >> 16) & 1u);          // round-to-nearest-even
  unsigned short s = (unsigned short)(u >> 16);
  return __builtin_bit_cast(bf16_t, s);
}
__device__ __forceinline__ float sigmoidf_(float x) {
  return 1.0f / (1.0f + __expf(-x));
}

// Async global -> LDS copy of 16 bytes per lane (CDNA5 GLOBAL_LOAD_ASYNC_TO_LDS_B128,
// tracked by ASYNCcnt). LDS byte offset = low 32 bits of the generic pointer.
__device__ __forceinline__ void async_ld_lds16(void* lds, const void* gaddr) {
  unsigned l = (unsigned)(uintptr_t)lds;
  asm volatile("global_load_async_to_lds_b128 %0, %1, off"
               :: "v"(l), "v"(gaddr) : "memory");
}
__device__ __forceinline__ void async_ld_lds16_off16(void* lds, const void* gaddr) {
  unsigned l = (unsigned)(uintptr_t)lds;
  asm volatile("global_load_async_to_lds_b128 %0, %1, off offset:16"
               :: "v"(l), "v"(gaddr) : "memory");
}
__device__ __forceinline__ void wait_asynccnt0() {
  asm volatile("s_wait_asynccnt 0" ::: "memory");
}

// A-fragment (16x32 bf16), contiguous k order matching B:
// lanes 0-15: K 0..15 ; lanes 16-31: K 16..31, M = lane&15.
__device__ __forceinline__ v16bf load_afrag(const bf16_t* rowp, int hi) {
  int kb = hi ? 16 : 0;
  v8bf lo = *(const v8bf*)(rowp + kb);
  v8bf hh = *(const v8bf*)(rowp + kb + 8);
  return __builtin_shufflevector(lo, hh, 0,1,2,3,4,5,6,7,8,9,10,11,12,13,14,15);
}

__device__ __forceinline__ v8f wmma_bf16(v16bf a, v16bf b, v8f c) {
  return __builtin_amdgcn_wmma_f32_16x16x32_bf16(false, a, false, b, (short)0, c, false, false);
}

// ---------------------------------------------------------------------------
// Kernel 1: convert/transpose weights to bf16, n-major [N][K]
// ---------------------------------------------------------------------------
__global__ void cvt_weights_kernel(const float* __restrict__ Wq, const float* __restrict__ Wk,
                                   const float* __restrict__ Wv, const float* __restrict__ Wo,
                                   const float* __restrict__ Wp,
                                   bf16_t* __restrict__ Wqkvn, bf16_t* __restrict__ Won,
                                   bf16_t* __restrict__ Wpn) {
  int idx = blockIdx.x * blockDim.x + threadIdx.x;
  const int QKV = NQKV * KQKV;
  const int OO  = DIMX * DIMX;
  if (idx < QKV) {
    int n = idx >> 9, k = idx & 511;
    float v;
    if (n < 512)       v = Wq[k * 512  + n];
    else if (n < 1536) v = Wk[k * 1024 + (n - 512)];
    else               v = Wv[k * 1024 + (n - 1536)];
    Wqkvn[idx] = f2bf(v);
  } else if (idx < QKV + OO) {
    int j = idx - QKV; int n = j >> 9, k = j & 511;
    Won[j] = f2bf(Wo[k * 512 + n]);
  } else if (idx < QKV + 2 * OO) {
    int j = idx - QKV - OO; int n = j >> 9, k = j & 511;
    Wpn[j] = f2bf(Wp[k * 512 + n]);
  }
}

// ---------------------------------------------------------------------------
// Kernel 2: LayerNorm + per-token gate dot products (lr / mom / dec)
// one block (256 threads) per token
// ---------------------------------------------------------------------------
__global__ __launch_bounds__(256) void ln_gates_kernel(
    const float* __restrict__ x, const unsigned char* __restrict__ mask,
    const float* __restrict__ ln_g, const float* __restrict__ ln_b,
    const float* __restrict__ w_lr, const float* __restrict__ b_lr,
    const float* __restrict__ w_mom, const float* __restrict__ b_mom,
    const float* __restrict__ w_dec, const float* __restrict__ b_dec,
    bf16_t* __restrict__ xnb, float* __restrict__ lr,
    float* __restrict__ momtok, float* __restrict__ dectok) {
  int t = blockIdx.x;
  int tid = threadIdx.x;
  const float* xr = x + (size_t)t * DIMX;
  float x0 = xr[tid], x1 = xr[tid + 256];

  __shared__ float red[256];
  red[tid] = x0 + x1; __syncthreads();
  for (int s = 128; s; s >>= 1) { if (tid < s) red[tid] += red[tid + s]; __syncthreads(); }
  float mu = red[0] * (1.0f / DIMX);
  __syncthreads();
  float d0 = x0 - mu, d1 = x1 - mu;
  red[tid] = d0 * d0 + d1 * d1; __syncthreads();
  for (int s = 128; s; s >>= 1) { if (tid < s) red[tid] += red[tid + s]; __syncthreads(); }
  float rstd = rsqrtf(red[0] * (1.0f / DIMX) + 1e-5f);

  float xn0 = d0 * rstd * ln_g[tid]       + ln_b[tid];
  float xn1 = d1 * rstd * ln_g[tid + 256] + ln_b[tid + 256];
  xnb[(size_t)t * DIMX + tid]       = f2bf(xn0);
  xnb[(size_t)t * DIMX + tid + 256] = f2bf(xn1);

  __shared__ float gr[12 * 256];
  for (int h = 0; h < 4; h++) {
    gr[(h    ) * 256 + tid] = xn0 * w_lr [tid * 4 + h] + xn1 * w_lr [(tid + 256) * 4 + h];
    gr[(4 + h) * 256 + tid] = xn0 * w_mom[tid * 4 + h] + xn1 * w_mom[(tid + 256) * 4 + h];
    gr[(8 + h) * 256 + tid] = xn0 * w_dec[tid * 4 + h] + xn1 * w_dec[(tid + 256) * 4 + h];
  }
  __syncthreads();
  for (int s = 128; s; s >>= 1) {
    if (tid < s) {
      for (int g = 0; g < 12; g++) gr[g * 256 + tid] += gr[g * 256 + tid + s];
    }
    __syncthreads();
  }
  if (tid < 4) {
    int h = tid;
    float m = mask[t] ? 1.0f : 0.0f;
    lr[t * 4 + h]     = sigmoidf_(gr[(h    ) * 256] + b_lr[h]) * m;
    momtok[t * 4 + h] = sigmoidf_(gr[(4 + h) * 256] + b_mom[h]);
    dectok[t * 4 + h] = sigmoidf_(gr[(8 + h) * 256] + b_dec[h]);
  }
}

// ---------------------------------------------------------------------------
// Kernel 3: chunk-mean of mom/dec gates -> (c,b,h)
// ---------------------------------------------------------------------------
__global__ void chunk_reduce_kernel(const float* __restrict__ momtok,
                                    const float* __restrict__ dectok,
                                    float* __restrict__ momc, float* __restrict__ decc) {
  int idx = blockIdx.x * blockDim.x + threadIdx.x;
  if (idx >= NCHUNK * BBX * HEADSX) return;
  int h = idx & 3;
  int b = (idx >> 2) & 3;
  int c = idx >> 4;
  float sm = 0.f, sd = 0.f;
  int base = (b * SSX + c * CHUNKX) * HEADSX + h;
  for (int cs = 0; cs < CHUNKX; cs++) {
    sm += momtok[base + cs * HEADSX];
    sd += dectok[base + cs * HEADSX];
  }
  momc[idx] = sm * (1.0f / CHUNKX);
  decc[idx] = sd * (1.0f / CHUNKX);
}

// ---------------------------------------------------------------------------
// Kernel 4/6/7: generic 128x128-tile bf16 WMMA GEMM, A row-major [M][K],
// Bn n-major [N][K]. Staging uses CDNA5 async global->LDS loads (ASYNCcnt).
// mode 0: scatter QKV, mode 1: bf16 out, mode 2: f32 +bias+resid
// ---------------------------------------------------------------------------
__global__ __launch_bounds__(256) void gemm_kernel(
    const bf16_t* __restrict__ A, const bf16_t* __restrict__ Bn, int K, int mode,
    bf16_t* __restrict__ oq, bf16_t* __restrict__ ok_, bf16_t* __restrict__ ov,
    float* __restrict__ outf, const float* __restrict__ bias,
    const float* __restrict__ resid) {
  __shared__ __align__(32) bf16_t As[128 * 32];
  __shared__ __align__(32) bf16_t Bs[128 * 32];
  int tid = threadIdx.x;
  int lane = tid & 31, wv = tid >> 5;
  int hi = (lane & 16) ? 1 : 0;
  int row0 = blockIdx.y * 128;
  int col0 = blockIdx.x * 128;

  v8f acc[8];
  for (int j = 0; j < 8; j++) for (int e = 0; e < 8; e++) acc[j][e] = 0.f;

  int sm = tid >> 1;            // staging row 0..127
  int sk = (tid & 1) * 16;      // staging k base {0,16}

  for (int kk = 0; kk < K; kk += 32) {
    __syncthreads();            // protect LDS reuse from previous iteration
    const bf16_t* ag = A  + (size_t)(row0 + sm) * K + kk + sk;
    const bf16_t* bg = Bn + (size_t)(col0 + sm) * K + kk + sk;
    // async global->LDS staging: 32 B per thread per tile, no VGPR round-trip
    async_ld_lds16      (As + sm * 32 + sk, ag);
    async_ld_lds16_off16(As + sm * 32 + sk, ag);
    async_ld_lds16      (Bs + sm * 32 + sk, bg);
    async_ld_lds16_off16(Bs + sm * 32 + sk, bg);
    wait_asynccnt0();
    __syncthreads();

    int ar = wv * 16 + (lane & 15);
    v16bf afrag = load_afrag(As + ar * 32, hi);
    int kb2 = hi ? 16 : 0;
    for (int j = 0; j < 8; j++) {
      int bc = j * 16 + (lane & 15);
      v16bf bfrag = *(const v16bf*)(Bs + bc * 32 + kb2);
      acc[j] = wmma_bf16(afrag, bfrag, acc[j]);
    }
  }

  for (int j = 0; j < 8; j++) {
    for (int i = 0; i < 8; i++) {
      int r = row0 + wv * 16 + i + (hi ? 8 : 0);
      int cc = col0 + j * 16 + (lane & 15);
      float val = acc[j][i];
      if (mode == 0) {
        if (cc < 512)       oq [(size_t)r * 512  + cc]          = f2bf(val);
        else if (cc < 1536) ok_[(size_t)r * 1024 + (cc - 512)]  = f2bf(val);
        else                ov [(size_t)r * 1024 + (cc - 1536)] = f2bf(val);
      } else if (mode == 1) {
        oq[(size_t)r * 512 + cc] = f2bf(val);
      } else {
        outf[(size_t)r * 512 + cc] = val + bias[cc] + resid[(size_t)r * 512 + cc];
      }
    }
  }
}

// ---------------------------------------------------------------------------
// Kernel 5: sequential chunk scan. grid = B*H, 256 threads (8 waves).
// M, S states (128x128 f32) live in registers in WMMA C-fragment layout.
// ---------------------------------------------------------------------------
__global__ __launch_bounds__(256) void scan_kernel(
    const bf16_t* __restrict__ qg, const bf16_t* __restrict__ kg,
    const bf16_t* __restrict__ vg, const float* __restrict__ lrg,
    const float* __restrict__ momc, const float* __restrict__ decc,
    bf16_t* __restrict__ memo) {
  __shared__ __align__(32) bf16_t Mb [128 * 128];  // M, col-major [e][d], bf16
  __shared__ __align__(32) bf16_t Kt [128 * 128];  // (k*lr)^T, row-major [d][n]
  __shared__ __align__(32) bf16_t Ebt[128 * 128];  // err, col-major [e][n]
  __shared__ float lrs[128];

  int tid = threadIdx.x, lane = tid & 31, wv = tid >> 5;
  int hi = (lane & 16) ? 1 : 0;
  int b = blockIdx.x >> 2, h = blockIdx.x & 3;

  // register-resident state: wave wv owns rows dd in [wv*16, wv*16+16)
  v8f Mreg[8], Sreg[8];
  for (int j = 0; j < 8; j++)
    for (int e = 0; e < 8; e++) { Mreg[j][e] = 0.f; Sreg[j][e] = 0.f; }

  for (int c = 0; c < NCHUNK; c++) {
    size_t tok0 = (size_t)b * SSX + (size_t)c * CHUNKX;

    // (a) spill M registers -> Mb (col-major bf16) ; load lr chunk
    for (int j = 0; j < 8; j++) {
      for (int i = 0; i < 8; i++) {
        int dd = wv * 16 + i + (hi ? 8 : 0);
        int e  = j * 16 + (lane & 15);
        Mb[e * 128 + dd] = f2bf(Mreg[j][i]);
      }
    }
    if (tid < 128) lrs[tid] = lrg[(tok0 + (tid >> 1)) * HEADSX + h];
    __syncthreads();

    // (b) build Kt[d][n] = k[n][d] * lr[n]
    for (int i = tid; i < 128 * 128; i += 256) {
      int n = i >> 7, d = i & 127;
      float kv = bf2f(kg[(tok0 + (n >> 1)) * 1024 + (h * 2 + (n & 1)) * 128 + d]);
      Kt[d * 128 + n] = f2bf(kv * lrs[n]);
    }
    float momv = momc[(c * BBX + b) * HEADSX + h];
    float decv = decc[(c * BBX + b) * HEADSX + h];
    __syncthreads();

    // (c) out = qc(64x128) @ M -> memo ; wave wv: rowtile wv>>1, coltiles (wv&1)*4..+3
    {
      int rt = wv >> 1;
      int cb = (wv & 1) * 4;
      v8f oacc[4];
      for (int j = 0; j < 4; j++) for (int e = 0; e < 8; e++) oacc[j][e] = 0.f;
      int arow = rt * 16 + (lane & 15);
      const bf16_t* ap = qg + (tok0 + arow) * 512 + h * 128;
      for (int kk = 0; kk < 128; kk += 32) {
        v16bf afrag = load_afrag(ap + kk, hi);
        for (int j = 0; j < 4; j++) {
          int e0 = (cb + j) * 16 + (lane & 15);
          v16bf bfrag = *(const v16bf*)(Mb + e0 * 128 + kk + (hi ? 16 : 0));
          oacc[j] = wmma_bf16(afrag, bfrag, oacc[j]);
        }
      }
      for (int j = 0; j < 4; j++) {
        for (int i = 0; i < 8; i++) {
          int cs = rt * 16 + i + (hi ? 8 : 0);
          int e  = (cb + j) * 16 + (lane & 15);
          memo[(tok0 + cs) * 512 + h * 128 + e] = f2bf(oacc[j][i]);
        }
      }
    }

    // (d) err = kc(128x128) @ M - vc -> Ebt (col-major bf16); wave wv: rowtile wv
    {
      v8f eacc[8];
      for (int j = 0; j < 8; j++) for (int e = 0; e < 8; e++) eacc[j][e] = 0.f;
      int arow = wv * 16 + (lane & 15);
      const bf16_t* ap = kg + (tok0 + (arow >> 1)) * 1024 + (h * 2 + (arow & 1)) * 128;
      for (int kk = 0; kk < 128; kk += 32) {
        v16bf afrag = load_afrag(ap + kk, hi);
        for (int j = 0; j < 8; j++) {
          int e0 = j * 16 + (lane & 15);
          v16bf bfrag = *(const v16bf*)(Mb + e0 * 128 + kk + (hi ? 16 : 0));
          eacc[j] = wmma_bf16(afrag, bfrag, eacc[j]);
        }
      }
      for (int j = 0; j < 8; j++) {
        for (int i = 0; i < 8; i++) {
          int n = wv * 16 + i + (hi ? 8 : 0);
          int e = j * 16 + (lane & 15);
          float vv = bf2f(vg[(tok0 + (n >> 1)) * 1024 + (h * 2 + (n & 1)) * 128 + e]);
          Ebt[e * 128 + n] = f2bf(eacc[j][i] - vv);
        }
      }
    }
    __syncthreads();

    // (e) grad = Kt @ err ; momentum + decay update on register state
    {
      v8f gacc[8];
      for (int j = 0; j < 8; j++) for (int e = 0; e < 8; e++) gacc[j][e] = 0.f;
      int arow = wv * 16 + (lane & 15);           // d
      const bf16_t* ap = Kt + arow * 128;
      for (int kk = 0; kk < 128; kk += 32) {
        v16bf afrag = load_afrag(ap + kk, hi);
        for (int j = 0; j < 8; j++) {
          int e0 = j * 16 + (lane & 15);
          v16bf bfrag = *(const v16bf*)(Ebt + e0 * 128 + kk + (hi ? 16 : 0));
          gacc[j] = wmma_bf16(afrag, bfrag, gacc[j]);
        }
      }
      float od = 1.0f - decv;
      for (int j = 0; j < 8; j++) {
        for (int i = 0; i < 8; i++) {
          float s = momv * Sreg[j][i] - gacc[j][i];
          Sreg[j][i] = s;
          Mreg[j][i] = od * Mreg[j][i] + s;
        }
      }
    }
    __syncthreads();
  }
}

// ---------------------------------------------------------------------------
extern "C" void kernel_launch(void* const* d_in, const int* in_sizes, int n_in,
                              void* d_out, int out_size, void* d_ws, size_t ws_size,
                              hipStream_t stream) {
  (void)in_sizes; (void)n_in; (void)out_size; (void)ws_size;
  const float* x            = (const float*)d_in[0];
  const unsigned char* mask = (const unsigned char*)d_in[1];
  const float* ln_g  = (const float*)d_in[2];
  const float* ln_b  = (const float*)d_in[3];
  const float* Wq    = (const float*)d_in[4];
  const float* Wk    = (const float*)d_in[5];
  const float* Wv    = (const float*)d_in[6];
  const float* w_lr  = (const float*)d_in[7];
  const float* b_lr  = (const float*)d_in[8];
  const float* w_mom = (const float*)d_in[9];
  const float* b_mom = (const float*)d_in[10];
  const float* w_dec = (const float*)d_in[11];
  const float* b_dec = (const float*)d_in[12];
  const float* Wo    = (const float*)d_in[13];
  const float* Wp    = (const float*)d_in[14];
  const float* bp    = (const float*)d_in[15];
  float* out = (float*)d_out;

  char* w = (char*)d_ws;
  auto take = [&](size_t bytes) -> char* {
    char* p = w; w += (bytes + 255) & ~(size_t)255; return p;
  };
  bf16_t* xnb   = (bf16_t*)take((size_t)NTOK * DIMX * 2);
  bf16_t* qb    = (bf16_t*)take((size_t)NTOK * 512  * 2);
  bf16_t* kb    = (bf16_t*)take((size_t)NTOK * 1024 * 2);
  bf16_t* vb    = (bf16_t*)take((size_t)NTOK * 1024 * 2);
  bf16_t* memo  = (bf16_t*)take((size_t)NTOK * 512  * 2);
  bf16_t* t1    = (bf16_t*)take((size_t)NTOK * 512  * 2);
  bf16_t* Wqkvn = (bf16_t*)take((size_t)NQKV * KQKV * 2);
  bf16_t* Won   = (bf16_t*)take((size_t)DIMX * DIMX * 2);
  bf16_t* Wpn   = (bf16_t*)take((size_t)DIMX * DIMX * 2);
  float*  lr    = (float*)take((size_t)NTOK * HEADSX * 4);
  float*  momtk = (float*)take((size_t)NTOK * HEADSX * 4);
  float*  dectk = (float*)take((size_t)NTOK * HEADSX * 4);
  float*  momc  = (float*)take((size_t)NCHUNK * BBX * HEADSX * 4);
  float*  decc  = (float*)take((size_t)NCHUNK * BBX * HEADSX * 4);

  int cvtTotal = NQKV * KQKV + 2 * DIMX * DIMX;
  cvt_weights_kernel<<<(cvtTotal + 255) / 256, 256, 0, stream>>>(
      Wq, Wk, Wv, Wo, Wp, Wqkvn, Won, Wpn);

  ln_gates_kernel<<<NTOK, 256, 0, stream>>>(
      x, mask, ln_g, ln_b, w_lr, b_lr, w_mom, b_mom, w_dec, b_dec,
      xnb, lr, momtk, dectk);

  chunk_reduce_kernel<<<(NCHUNK * BBX * HEADSX + 255) / 256, 256, 0, stream>>>(
      momtk, dectk, momc, decc);

  gemm_kernel<<<dim3(NQKV / 128, NTOK / 128), 256, 0, stream>>>(
      xnb, Wqkvn, KQKV, 0, qb, kb, vb, nullptr, nullptr, nullptr);

  scan_kernel<<<BBX * HEADSX, 256, 0, stream>>>(qb, kb, vb, lr, momc, decc, memo);

  gemm_kernel<<<dim3(512 / 128, NTOK / 128), 256, 0, stream>>>(
      memo, Won, 512, 1, t1, nullptr, nullptr, nullptr, nullptr, nullptr);

  gemm_kernel<<<dim3(512 / 128, NTOK / 128), 256, 0, stream>>>(
      t1, Wpn, 512, 2, nullptr, nullptr, nullptr, out, bp, x);
}
